// RoPEGlobalAttention_61065845015295
// MI455X (gfx1250) — matrix operand order repaired and checked by reading, not compiled, not run on hardware
//
#include <hip/hip_runtime.h>
#include <math.h>

typedef __attribute__((ext_vector_type(16))) __bf16 v16bf;
typedef __attribute__((ext_vector_type(8)))  float  v8f;

#define DIM       512
#define NB        8
#define NHEADS    8
#define HD        64
#define NTOK      1024
#define NROWS     (NB * NTOK)          // 8192
#define QKV_N     (3 * DIM)            // 1536
#define HEAD_ELEMS 65536               // 1024*64 (q/k) == 64*1024 (vt)

// log2(10000)/32 : freqs[i] = 2^(-i * RO_C)
#define RO_C   0.41524101186092025f
#define LOG2E  1.4426950408889634f

static __device__ __forceinline__ v16bf load2x16(const __bf16* p0, const __bf16* p1) {
  union { v16bf v; uint4 u[2]; } t;
  t.u[0] = *(const uint4*)p0;
  t.u[1] = *(const uint4*)p1;
  return t.v;
}

// LDS byte offset of a __shared__ object (AS3 pointers are offsets from LDS base)
static __device__ __forceinline__ uint32_t lds_off(const void* p) {
  return (uint32_t)(uintptr_t)(__attribute__((address_space(3))) const char*)p;
}

// CDNA5 async DMA: global -> LDS, 16B per lane, tracked by ASYNCcnt
static __device__ __forceinline__ void g2lds_b128(uint32_t lds_byte, const __bf16* g) {
  asm volatile("global_load_async_to_lds_b128 %0, %1, off"
               :: "v"(lds_byte), "v"(g) : "memory");
}

// ---------------------------------------------------------------- w -> bf16
__global__ void cvt_w_kernel(const float* __restrict__ w, __bf16* __restrict__ wb) {
  int i = blockIdx.x * 256 + threadIdx.x;
  if (i < QKV_N * DIM) wb[i] = (__bf16)w[i];
}

// ------------------------------------------------- LayerNorm + cast to bf16
__global__ void __launch_bounds__(256)
ln_kernel(const float* __restrict__ x, const float* __restrict__ gamma,
          const float* __restrict__ beta, __bf16* __restrict__ xn) {
  __shared__ float xs[16][513];
  __shared__ float ps[16][17];
  __shared__ float pq[16][17];
  __shared__ float s_mean[16], s_inv[16];

  const int tid  = threadIdx.x;
  const int nloc = tid & 15;
  const int cg   = tid >> 4;             // 16 groups x 32 channels
  const int m0   = blockIdx.x * 16;      // global row base
  const int b    = m0 >> 10;
  const int n0   = m0 & 1023;

  float sum = 0.f, sq = 0.f;
  const size_t xbase = ((size_t)b * DIM) << 10;
  for (int j = 0; j < 32; ++j) {
    int c = cg * 32 + j;
    float v = x[xbase + ((size_t)c << 10) + n0 + nloc];
    xs[nloc][c] = v;
    sum += v; sq += v * v;
  }
  ps[nloc][cg] = sum;
  pq[nloc][cg] = sq;
  __syncthreads();

  if (tid < 16) {
    float s = 0.f, q = 0.f;
    for (int g = 0; g < 16; ++g) { s += ps[tid][g]; q += pq[tid][g]; }
    float mean = s * (1.0f / DIM);
    float var  = q * (1.0f / DIM) - mean * mean;
    s_mean[tid] = mean;
    s_inv[tid]  = rsqrtf(var + 1e-5f);
  }
  __syncthreads();

  for (int kidx = 0; kidx < 32; ++kidx) {
    int flat = kidx * 256 + tid;
    int c = flat & 511;
    int n = flat >> 9;
    float v = (xs[n][c] - s_mean[n]) * s_inv[n] * gamma[c] + beta[c];
    xn[(size_t)(m0 + n) * DIM + c] = (__bf16)v;
  }
}

// ---------------- QKV GEMM (bf16 WMMA, f32 acc) + bias + RoPE + V transpose
// Block: 8 waves; wave computes 16(M) x 64(N). Weight K-slices staged in LDS
// via async DMA: 4-deep ring, issue distance 2, ONE barrier per stage.
#define BPAD 40   // 32 data + 8 pad bf16 per LDS row (20-word stride, conflict-free)

__global__ void __launch_bounds__(256)
qkv_gemm_rope_kernel(const __bf16* __restrict__ xn, const __bf16* __restrict__ wb,
                     const float* __restrict__ bias,
                     __bf16* __restrict__ qbuf, __bf16* __restrict__ kbuf,
                     __bf16* __restrict__ vt) {
  __shared__ __align__(16) __bf16 bls[4][64 * BPAD];   // 4 x 5120 B

  const int tid  = threadIdx.x;
  const int lane = tid & 31;
  const int wave = tid >> 5;
  const int l15  = lane & 15;
  const int hi   = lane >> 4;

  const int mblk  = blockIdx.x & 63;
  const int nblk  = blockIdx.x >> 6;           // 0..23
  const int mtile = mblk * 128 + wave * 16;
  const int jb    = nblk * 64;

  // async-issue coordinates: 64 rows x 4 chunks of 16B
  const int brow = tid >> 2;
  const int bch  = tid & 3;
  const __bf16* bsrc = wb + (size_t)(jb + brow) * DIM + bch * 8;
  uint32_t bdst[4];
  for (int i = 0; i < 4; ++i)
    bdst[i] = lds_off(&bls[i][brow * BPAD + bch * 8]);

  v8f acc[4];
  for (int t = 0; t < 4; ++t)
    for (int v = 0; v < 8; ++v) acc[t][v] = 0.f;

  const __bf16* aptr = xn + (size_t)(mtile + l15) * DIM + hi * 8;

  // prologue: stage K-slices 0 and 1
  g2lds_b128(bdst[0], bsrc);
  g2lds_b128(bdst[1], bsrc + 32);

#pragma unroll
  for (int kk = 0; kk < 16; ++kk) {
    const int cur = kk & 3;
    if (kk + 2 < 16)
      g2lds_b128(bdst[(kk + 2) & 3], bsrc + (kk + 2) * 32);
    if (kk <= 13)      asm volatile("s_wait_asynccnt 0x2" ::: "memory");
    else if (kk == 14) asm volatile("s_wait_asynccnt 0x1" ::: "memory");
    else               asm volatile("s_wait_asynccnt 0x0" ::: "memory");
    __syncthreads();   // single barrier per stage (WAR covered by 4-deep ring)

    const int k0 = kk * 32;
    __builtin_prefetch(aptr + k0 + 64, 0, 3);
    v16bf a = load2x16(aptr + k0, aptr + k0 + 16);
    const __bf16* bl = &bls[cur][hi * 8];
#pragma unroll
    for (int t = 0; t < 4; ++t) {
      const __bf16* bp = bl + (t * 16 + l15) * BPAD;
      v16bf bmat = load2x16(bp, bp + 16);
      acc[t] = __builtin_amdgcn_wmma_f32_16x16x32_bf16(
          false, a, false, bmat, (short)0, acc[t], false, false);
    }
  }

  // bias
  float av[4][8];
  for (int t = 0; t < 4; ++t) {
    float bia = bias[jb + t * 16 + l15];
    for (int v = 0; v < 8; ++v) av[t][v] = acc[t][v] + bia;
  }

  const int sect = jb >> 9;                    // 0=q 1=k 2=v
  const int h    = (jb >> 6) & 7;
  const int nb   = mtile & 1023;
  const int b    = mtile >> 10;
  const size_t headoff = ((size_t)(b * NHEADS + h)) * HEAD_ELEMS;

  if (sect == 2) {
    for (int t = 0; t < 4; ++t) {
      int d = t * 16 + l15;
      union { __bf16 e[8]; uint4 u; } pk;
      for (int v = 0; v < 8; ++v) pk.e[v] = (__bf16)av[t][v];
      __bf16* p = vt + headoff + ((size_t)d << 10) + nb + 8 * hi;
      *(uint4*)p = pk.u;
    }
  } else {
    __bf16* dst = (sect == 0) ? qbuf : kbuf;
    // Q gets hd^-0.5 and log2(e) folded in (softmax done in exp2 domain)
    const float qs = (sect == 0) ? (0.125f * LOG2E) : 1.0f;
    for (int t = 0; t < 2; ++t) {
      int d_lo = t * 16 + l15;                      // 0..31
      int i_lo = d_lo >> 1;
      float f_lo = exp2f(-(float)i_lo * RO_C);
      float f_hi = exp2f(-(float)(i_lo + 16) * RO_C);
      for (int v = 0; v < 8; ++v) {
        int   n   = nb + 8 * hi + v;
        float pos = (float)n;
        float s1, c1, s2, c2;
        __sincosf(pos * f_lo, &s1, &c1);
        __sincosf(pos * f_hi, &s2, &c2);
        float lo = (av[t][v]     * c1 - av[t + 2][v] * s1) * qs;
        float hg = (av[t + 2][v] * c2 + av[t][v]     * s2) * qs;
        __bf16* p = dst + headoff + ((size_t)n << 6);
        p[d_lo]      = (__bf16)lo;
        p[d_lo + 32] = (__bf16)hg;
      }
    }
  }
}

// ------------------------------------- flash attention (bf16 WMMA, f32 acc)
// Block: 8 waves; wave owns 16 query rows, full d=64. 32 keys/iter.
// K/V tiles: 4-deep async LDS ring, issue distance 2, one barrier per stage.
// Softmax in exp2 domain; row-sum lane-reduction deferred to after the loop.
#define KPAD 72   // 64 data + 8 pad bf16 (36-word stride)
#define VPAD 40   // 32 data + 8 pad bf16 (20-word stride)

__global__ void __launch_bounds__(256)
attn_kernel(const __bf16* __restrict__ q, const __bf16* __restrict__ k,
            const __bf16* __restrict__ vt, float* __restrict__ out) {
  __shared__ __align__(16) __bf16 pbuf[8][16 * 32];     //  8192 B (per-wave P)
  __shared__ __align__(16) __bf16 klds[4][32 * KPAD];   // 18432 B
  __shared__ __align__(16) __bf16 vlds[4][64 * VPAD];   // 20480 B

  const int tid  = threadIdx.x;
  const int lane = tid & 31;
  const int wave = tid >> 5;
  const int l15  = lane & 15;
  const int hi   = lane >> 4;

  const int bh    = blockIdx.x >> 3;           // 0..63
  const int qblk  = blockIdx.x & 7;
  const int qbase = qblk * 128 + wave * 16;

  const size_t hoff = (size_t)bh * HEAD_ELEMS;
  const __bf16* qrow = q + hoff + (size_t)(qbase + l15) * HD + hi * 8;
  const __bf16* kh   = k + hoff;
  const __bf16* vh   = vt + hoff;
  __bf16* pw = pbuf[wave];

  // async-issue coordinates
  const int krow = tid >> 3, kch = tid & 7;    // 32 rows x 8 chunks (K)
  const int vrow = tid >> 2, vch = tid & 3;    // 64 rows x 4 chunks (V)
  const __bf16* ksrc = kh + (size_t)krow * HD + kch * 8;
  const __bf16* vsrc = vh + ((size_t)vrow << 10) + vch * 8;
  uint32_t kdst[4], vdst[4];
  for (int i = 0; i < 4; ++i) {
    kdst[i] = lds_off(&klds[i][krow * KPAD + kch * 8]);
    vdst[i] = lds_off(&vlds[i][vrow * VPAD + vch * 8]);
  }

  v16bf qa[2];
  for (int ks = 0; ks < 2; ++ks)
    qa[ks] = load2x16(qrow + ks * 32, qrow + ks * 32 + 16);

  float rmax[8], rsum[8];
  v8f o[4];
  for (int v = 0; v < 8; ++v) { rmax[v] = -INFINITY; rsum[v] = 0.f; }
  for (int t = 0; t < 4; ++t)
    for (int v = 0; v < 8; ++v) o[t][v] = 0.f;

  // prologue: stage key-tiles 0 and 1
  g2lds_b128(kdst[0], ksrc);
  g2lds_b128(vdst[0], vsrc);
  g2lds_b128(kdst[1], ksrc + (size_t)32 * HD);
  g2lds_b128(vdst[1], vsrc + 32);

#pragma unroll 2
  for (int kt = 0; kt < 32; ++kt) {            // 32 iters x 32 keys = 1024
    const int cur = kt & 3;
    if (kt + 2 < 32) {
      const int nk = (kt + 2) * 32;
      g2lds_b128(kdst[(kt + 2) & 3], ksrc + (size_t)nk * HD);
      g2lds_b128(vdst[(kt + 2) & 3], vsrc + nk);
    }
    if (kt <= 29)      asm volatile("s_wait_asynccnt 0x4" ::: "memory");
    else if (kt == 30) asm volatile("s_wait_asynccnt 0x2" ::: "memory");
    else               asm volatile("s_wait_asynccnt 0x0" ::: "memory");
    __syncthreads();   // single barrier per stage (WAR covered by 4-deep ring)

    // S = Q * K^T  (2 col-tiles x 2 K-steps) from LDS
    v8f s[2];
    const __bf16* kl = &klds[cur][hi * 8];
#pragma unroll
    for (int ct = 0; ct < 2; ++ct) {
      for (int v = 0; v < 8; ++v) s[ct][v] = 0.f;
      const __bf16* kp = kl + (ct * 16 + l15) * KPAD;
#pragma unroll
      for (int ks = 0; ks < 2; ++ks) {
        v16bf kb = load2x16(kp + ks * 32, kp + ks * 32 + 16);
        s[ct] = __builtin_amdgcn_wmma_f32_16x16x32_bf16(
            false, qa[ks], false, kb, (short)0, s[ct], false, false);
      }
    }

    // online softmax per row; corr is row-uniform so the running sum can stay
    // per-lane (cross-lane sum reduction deferred to after the loop)
#pragma unroll
    for (int v = 0; v < 8; ++v) {
      float a0 = s[0][v], a1 = s[1][v];
      float mx = fmaxf(a0, a1);
      mx = fmaxf(mx, __shfl_xor(mx, 1, 32));
      mx = fmaxf(mx, __shfl_xor(mx, 2, 32));
      mx = fmaxf(mx, __shfl_xor(mx, 4, 32));
      mx = fmaxf(mx, __shfl_xor(mx, 8, 32));
      float mn   = fmaxf(rmax[v], mx);
      float corr = exp2f(rmax[v] - mn);
      float p0 = exp2f(a0 - mn);
      float p1 = exp2f(a1 - mn);
      rsum[v] = rsum[v] * corr + (p0 + p1);    // per-lane partial
      rmax[v] = mn;
      o[0][v] *= corr; o[1][v] *= corr; o[2][v] *= corr; o[3][v] *= corr;
      int r = v + 8 * hi;
      pw[r * 32 + l15]      = (__bf16)p0;
      pw[r * 32 + 16 + l15] = (__bf16)p1;
    }
    __builtin_amdgcn_wave_barrier();
    asm volatile("" ::: "memory");             // LDS in-order per wave

    // reload P in A-operand layout
    const __bf16* pp = pw + l15 * 32 + hi * 8;
    v16bf pa = load2x16(pp, pp + 16);
    __builtin_amdgcn_wave_barrier();
    asm volatile("" ::: "memory");

    // O += P * V   (V tile in LDS: d-major, key-contiguous)
    const __bf16* vl = &vlds[cur][hi * 8];
#pragma unroll
    for (int dt = 0; dt < 4; ++dt) {
      const __bf16* vp = vl + (dt * 16 + l15) * VPAD;
      v16bf vb = load2x16(vp, vp + 16);
      o[dt] = __builtin_amdgcn_wmma_f32_16x16x32_bf16(
          false, pa, false, vb, (short)0, o[dt], false, false);
    }
  }

  // finalize row sums across the 16 lanes of each half-wave
#pragma unroll
  for (int v = 0; v < 8; ++v) {
    float s = rsum[v];
    s += __shfl_xor(s, 1, 32);
    s += __shfl_xor(s, 2, 32);
    s += __shfl_xor(s, 4, 32);
    s += __shfl_xor(s, 8, 32);
    rsum[v] = s;
  }

  // normalize + write out[b, c, n]  (8 consecutive n per lane -> float4 x2)
  const int b = bh >> 3, h = bh & 7;
#pragma unroll
  for (int t = 0; t < 4; ++t) {
    int c = h * HD + t * 16 + l15;
    float* p = out + (((size_t)(b * DIM + c)) << 10) + qbase + 8 * hi;
    float4 f0, f1;
    f0.x = o[t][0] / rsum[0]; f0.y = o[t][1] / rsum[1];
    f0.z = o[t][2] / rsum[2]; f0.w = o[t][3] / rsum[3];
    f1.x = o[t][4] / rsum[4]; f1.y = o[t][5] / rsum[5];
    f1.z = o[t][6] / rsum[6]; f1.w = o[t][7] / rsum[7];
    *(float4*)p       = f0;
    *((float4*)p + 1) = f1;
  }
}

// --------------------------------------------------------------------------
extern "C" void kernel_launch(void* const* d_in, const int* in_sizes, int n_in,
                              void* d_out, int out_size, void* d_ws, size_t ws_size,
                              hipStream_t stream) {
  const float* x     = (const float*)d_in[0];
  const float* w_qkv = (const float*)d_in[1];
  const float* b_qkv = (const float*)d_in[2];
  const float* gamma = (const float*)d_in[3];
  const float* beta  = (const float*)d_in[4];
  float* out = (float*)d_out;

  char* ws = (char*)d_ws;
  __bf16* xn = (__bf16*)ws;                         //  8,388,608 B
  __bf16* wb = (__bf16*)(ws + 8388608);             //  1,572,864 B
  __bf16* qb = (__bf16*)(ws + 9961472);             //  8,388,608 B
  __bf16* kb = (__bf16*)(ws + 18350080);            //  8,388,608 B
  __bf16* vt = (__bf16*)(ws + 26738688);            //  8,388,608 B (end ~35.1MB)

  cvt_w_kernel<<<(QKV_N * DIM + 255) / 256, 256, 0, stream>>>(w_qkv, wb);
  ln_kernel<<<NROWS / 16, 256, 0, stream>>>(x, gamma, beta, xn);
  qkv_gemm_rope_kernel<<<64 * 24, 256, 0, stream>>>(xn, wb, b_qkv, qb, kb, vt);
  attn_kernel<<<64 * 8, 256, 0, stream>>>(qb, kb, vt, out);
}
